// DGCNGRU_65206193487970
// MI455X (gfx1250) — matrix operand order repaired and checked by reading, not compiled, not run on hardware
//
#include <hip/hip_runtime.h>

#define N_MESS 50000
#define MAX_NEI 8
#define H 128
#define MSG_PER_BLK 32
#define NPAD 50016                       // 1563 * 32, also 3126 * 16
#define NBLK_STEP (NPAD / MSG_PER_BLK)   // 1563
#define NBLK_R1   (NPAD / 16)            // 3126

typedef __bf16 bf16_t;
typedef __attribute__((ext_vector_type(16))) __bf16 v16bf;
typedef __attribute__((ext_vector_type(8)))  __bf16 v8bf;
typedef __attribute__((ext_vector_type(8)))  float  v8f;

union V16U { v16bf v; v8bf h[2]; };

// A-matrix fragment (16x32 bf16, MxK), contiguous K to match B:
// lanes 0-15: row M=lane, elems 0-15 = K 0..15
// lanes 16-31: row M=lane-16, elems 0-15 = K 16..31
__device__ __forceinline__ v16bf load_a_frag(const bf16_t* base, int stride,
                                             int m_base, int k_base, int lane) {
  const int hi = lane >> 4;
  const int m  = m_base + (lane & 15);
  const bf16_t* p = base + (size_t)m * stride + k_base + hi * 16;
  V16U u;
  u.h[0] = *(const v8bf*)(p);
  u.h[1] = *(const v8bf*)(p + 8);
  return u.v;
}

// B-matrix fragment (32x16 bf16, KxN) from row-major weight W[out][in]:
// lane holds column N = n_base+(lane&15); K contiguous along W's row.
// lanes 0-15: K 0..15, lanes 16-31: K 16..31 -> one 32B load per lane.
__device__ __forceinline__ v16bf load_b_frag(const bf16_t* w, int stride,
                                             int n_base, int k_base, int lane) {
  const int hi = lane >> 4;
  const int n  = n_base + (lane & 15);
  const bf16_t* p = w + (size_t)n * stride + k_base + hi * 16;
  return *(const v16bf*)(p);
}

__device__ __forceinline__ v8f wmma_bf16(v16bf a, v16bf b, v8f c) {
  return __builtin_amdgcn_wmma_f32_16x16x32_bf16(false, a, false, b,
                                                 (short)0, c, false, false);
}

__device__ __forceinline__ float sigmoidf_(float x) {
  return 1.f / (1.f + __expf(-x));
}

// Generic->LDS byte address: flat LDS aperture uses addr[31:0] as the LDS
// offset (ISA 10.2 aperture mapping), which is what GLOBAL_LOAD_ASYNC_TO_LDS
// expects in VDST.
__device__ __forceinline__ unsigned lds_off(const void* p) {
  return (unsigned)(uintptr_t)p;
}

// Async memory->LDS copies (CDNA5 GLOBAL_LOAD_ASYNC_TO_LDS, ASYNCcnt).
// INST_OFFSET is added to both the global and LDS addresses.
#define GLD_ASYNC16(OFF)                                                   \
  asm volatile("global_load_async_to_lds_b128 %0, %1, off offset:" #OFF    \
               :: "v"(ldsa), "v"(ga) : "memory")

__device__ __forceinline__ void async_copy_256B(unsigned ldsa, uint64_t ga) {
  GLD_ASYNC16(0);   GLD_ASYNC16(16);  GLD_ASYNC16(32);  GLD_ASYNC16(48);
  GLD_ASYNC16(64);  GLD_ASYNC16(80);  GLD_ASYNC16(96);  GLD_ASYNC16(112);
  GLD_ASYNC16(128); GLD_ASYNC16(144); GLD_ASYNC16(160); GLD_ASYNC16(176);
  GLD_ASYNC16(192); GLD_ASYNC16(208); GLD_ASYNC16(224); GLD_ASYNC16(240);
}

__device__ __forceinline__ void async_copy_128B(unsigned ldsa, uint64_t ga) {
  GLD_ASYNC16(0);   GLD_ASYNC16(16);  GLD_ASYNC16(32);  GLD_ASYNC16(48);
  GLD_ASYNC16(64);  GLD_ASYNC16(80);  GLD_ASYNC16(96);  GLD_ASYNC16(112);
}

__device__ __forceinline__ void async_wait0() {
  asm volatile("s_wait_asynccnt 0x0" ::: "memory");
}

// ---- prep: fp32 -> bf16 casts of fmess + weights, zero initial h ----------
__global__ void DGCNGRU_prep(const float* __restrict__ fmess,
                             const float* __restrict__ Wz,
                             const float* __restrict__ Wr,
                             const float* __restrict__ Ur,
                             const float* __restrict__ Wh,
                             bf16_t* __restrict__ fm_bf,
                             bf16_t* __restrict__ h0,
                             bf16_t* __restrict__ Wz_bf,
                             bf16_t* __restrict__ Wr_bf,
                             bf16_t* __restrict__ Ur_bf,
                             bf16_t* __restrict__ Wh_bf) {
  const int i = blockIdx.x * blockDim.x + threadIdx.x;
  if (i < NPAD * H) {
    fm_bf[i] = (i < N_MESS * H) ? (bf16_t)fmess[i] : (bf16_t)0.f;
    h0[i]    = (bf16_t)0.f;
  }
  if (i < 128 * 256) {
    Wz_bf[i] = (bf16_t)Wz[i];
    Wh_bf[i] = (bf16_t)Wh[i];
  }
  if (i < 128 * 128) {
    Wr_bf[i] = (bf16_t)Wr[i];
    Ur_bf[i] = (bf16_t)Ur[i];
  }
}

// ---- r1 = fmess @ W_r.T  (loop-invariant, computed once) ------------------
__global__ void DGCNGRU_r1(const bf16_t* __restrict__ fm_bf,
                           const bf16_t* __restrict__ Wr_bf,
                           float* __restrict__ r1) {
  const int lane = threadIdx.x & 31;
  const int wave = threadIdx.x >> 5;   // 0..7 -> N-tile
  const int hi   = lane >> 4;
  const int msg_base = blockIdx.x * 16;

  v16bf a[4];
#pragma unroll
  for (int kt = 0; kt < 4; kt++)
    a[kt] = load_a_frag(fm_bf, H, msg_base, kt * 32, lane);

  v8f c = {};
#pragma unroll
  for (int kt = 0; kt < 4; kt++) {
    v16bf b = load_b_frag(Wr_bf, H, wave * 16, kt * 32, lane);
    c = wmma_bf16(a[kt], b, c);
  }
  const int col = wave * 16 + (lane & 15);
#pragma unroll
  for (int j = 0; j < 8; j++) {
    const int m = msg_base + j + hi * 8;
    r1[(size_t)m * H + col] = c[j];
  }
}

// ---- one GRU depth step: 32 messages / block ------------------------------
__global__ void DGCNGRU_step(const bf16_t* __restrict__ h_in,
                             bf16_t* __restrict__ h_out,
                             const int* __restrict__ bgraph,
                             const bf16_t* __restrict__ fm_bf,
                             const float* __restrict__ r1,
                             const bf16_t* __restrict__ Wz_bf,
                             const bf16_t* __restrict__ Ur_bf,
                             const bf16_t* __restrict__ Wh_bf,
                             const float* __restrict__ b_z,
                             const float* __restrict__ b_Ur,
                             const float* __restrict__ b_h,
                             float* __restrict__ out_f32,
                             int write_out) {
  __shared__ __align__(32) bf16_t hnei[256 * H];            // 64 KB gather tile
  __shared__ __align__(32) bf16_t Ur_l[128 * H];            // 32 KB staged U_r
  __shared__ __align__(32) float  sumh[MSG_PER_BLK * H];    // 16 KB
  __shared__ __align__(32) bf16_t sumh_bf[MSG_PER_BLK * H]; //  8 KB
  __shared__ __align__(32) bf16_t sumg_bf[MSG_PER_BLK * H]; //  8 KB

  const int tid  = threadIdx.x;
  const int lane = tid & 31;
  const int wave = tid >> 5;    // 0..7
  const int hi   = lane >> 4;
  const int msg_base = blockIdx.x * MSG_PER_BLK;

  // ---- async gather h_nei (256 rows x 256B) + stage U_r (32KB), mem->LDS --
  {
    const int row  = tid;               // one 256B row per thread
    const int msgl = row >> 3;
    const int k    = row & 7;
    const int msgG = msg_base + msgl;
    const int nb   = (msgG < N_MESS) ? bgraph[msgG * MAX_NEI + k] : 0;
    async_copy_256B(lds_off(&hnei[row * H]),
                    (uint64_t)(uintptr_t)(h_in + (size_t)nb * H));
    // 128B of U_r per thread
    async_copy_128B(lds_off(&Ur_l[tid * 64]),
                    (uint64_t)(uintptr_t)(Ur_bf + (size_t)tid * 64));
  }
  async_wait0();
  __syncthreads();

  // ---- sum_h: wave w owns messages 4w..4w+3 -------------------------------
  {
    const int c0 = lane & 15;
#pragma unroll
    for (int s = 0; s < 2; s++) {
      const int msgl = 4 * wave + 2 * s + hi;
#pragma unroll
      for (int c8 = 0; c8 < 8; c8++) {
        const int col = c0 + c8 * 16;
        float sum = 0.f;
#pragma unroll
        for (int k = 0; k < 8; k++) sum += (float)hnei[(msgl * 8 + k) * H + col];
        sumh[msgl * H + col]    = sum;
        sumh_bf[msgl * H + col] = (bf16_t)sum;
      }
    }
  }

  // ---- r2 = h_nei @ U_r.T ; r = sigmoid(r1+r2+b) ; lane-local gated reduce.
  // Wave w owns A row-blocks 2w,2w+1 (rows 32w..32w+31); each B fragment
  // (from LDS) is reused across both row-blocks.
  {
    v16bf a[2][4];
#pragma unroll
    for (int s = 0; s < 2; s++)
#pragma unroll
      for (int kt = 0; kt < 4; kt++)
        a[s][kt] = load_a_frag(hnei, H, (2 * wave + s) * 16, kt * 32, lane);

#pragma unroll
    for (int nt = 0; nt < 8; nt++) {
      v16bf b[4];
#pragma unroll
      for (int kt = 0; kt < 4; kt++)
        b[kt] = load_b_frag(Ur_l, H, nt * 16, kt * 32, lane);

      const int col = nt * 16 + (lane & 15);
      const float bur = b_Ur[col];
#pragma unroll
      for (int s = 0; s < 2; s++) {
        v8f c = {};
#pragma unroll
        for (int kt = 0; kt < 4; kt++) c = wmma_bf16(a[s][kt], b[kt], c);

        const int rb   = 2 * wave + s;
        const int msgl = 2 * rb + hi;
        const float r1v = r1[(size_t)(msg_base + msgl) * H + col];
        float sg = 0.f;
#pragma unroll
        for (int j = 0; j < 8; j++) {
          const float rr = sigmoidf_(r1v + c[j] + bur);
          const float hv = (float)hnei[(16 * rb + hi * 8 + j) * H + col];
          sg += rr * hv;
        }
        sumg_bf[msgl * H + col] = (bf16_t)sg;
      }
    }
  }
  __syncthreads();

  // ---- z / pre_h GEMMs: [32,256] @ W.T; wave w -> N-tile w, B reused over
  // both message-tiles.
  {
    v16bf af[2][4], as[2][4];
#pragma unroll
    for (int mt = 0; mt < 2; mt++)
#pragma unroll
      for (int kt = 0; kt < 4; kt++) {
        af[mt][kt] = load_a_frag(fm_bf, H, msg_base + mt * 16, kt * 32, lane);
        as[mt][kt] = load_a_frag(sumh_bf, H, mt * 16, kt * 32, lane);
      }

    v8f cz[2] = {{}, {}};
#pragma unroll
    for (int kt = 0; kt < 4; kt++) {
      v16bf b = load_b_frag(Wz_bf, 256, wave * 16, kt * 32, lane);
#pragma unroll
      for (int mt = 0; mt < 2; mt++) cz[mt] = wmma_bf16(af[mt][kt], b, cz[mt]);
    }
#pragma unroll
    for (int kt = 0; kt < 4; kt++) {
      v16bf b = load_b_frag(Wz_bf, 256, wave * 16, 128 + kt * 32, lane);
#pragma unroll
      for (int mt = 0; mt < 2; mt++) cz[mt] = wmma_bf16(as[mt][kt], b, cz[mt]);
    }

#pragma unroll
    for (int mt = 0; mt < 2; mt++)
#pragma unroll
      for (int kt = 0; kt < 4; kt++)
        as[mt][kt] = load_a_frag(sumg_bf, H, mt * 16, kt * 32, lane);

    v8f ch[2] = {{}, {}};
#pragma unroll
    for (int kt = 0; kt < 4; kt++) {
      v16bf b = load_b_frag(Wh_bf, 256, wave * 16, kt * 32, lane);
#pragma unroll
      for (int mt = 0; mt < 2; mt++) ch[mt] = wmma_bf16(af[mt][kt], b, ch[mt]);
    }
#pragma unroll
    for (int kt = 0; kt < 4; kt++) {
      v16bf b = load_b_frag(Wh_bf, 256, wave * 16, 128 + kt * 32, lane);
#pragma unroll
      for (int mt = 0; mt < 2; mt++) ch[mt] = wmma_bf16(as[mt][kt], b, ch[mt]);
    }

    const int col = wave * 16 + (lane & 15);
    const float bz = b_z[col], bh = b_h[col];
#pragma unroll
    for (int mt = 0; mt < 2; mt++) {
#pragma unroll
      for (int j = 0; j < 8; j++) {
        const int m  = mt * 16 + j + hi * 8;
        const int mG = msg_base + m;
        const float z  = sigmoidf_(cz[mt][j] + bz);
        const float p  = tanhf(ch[mt][j] + bh);
        const float sh = sumh[m * H + col];
        float hnew = (1.f - z) * sh + z * p;
        if (mG == 0) hnew = 0.f;   // message 0 is padding
        h_out[(size_t)mG * H + col] = (bf16_t)hnew;
        if (write_out && mG < N_MESS) out_f32[(size_t)mG * H + col] = hnew;
      }
    }
  }
}

extern "C" void kernel_launch(void* const* d_in, const int* in_sizes, int n_in,
                              void* d_out, int out_size, void* d_ws, size_t ws_size,
                              hipStream_t stream) {
  const float* fmess  = (const float*)d_in[0];
  const int*   bgraph = (const int*)d_in[1];
  const float* W_z    = (const float*)d_in[2];
  const float* b_z    = (const float*)d_in[3];
  const float* W_r    = (const float*)d_in[4];
  const float* U_r    = (const float*)d_in[5];
  const float* b_Ur   = (const float*)d_in[6];
  const float* W_h    = (const float*)d_in[7];
  const float* b_h    = (const float*)d_in[8];
  float* out = (float*)d_out;

  char* p = (char*)d_ws;
  bf16_t* fm_bf = (bf16_t*)p; p += (size_t)NPAD * H * sizeof(bf16_t);
  bf16_t* hA    = (bf16_t*)p; p += (size_t)NPAD * H * sizeof(bf16_t);
  bf16_t* hB    = (bf16_t*)p; p += (size_t)NPAD * H * sizeof(bf16_t);
  float*  r1    = (float*)p;  p += (size_t)NPAD * H * sizeof(float);
  bf16_t* Wz_bf = (bf16_t*)p; p += (size_t)128 * 256 * sizeof(bf16_t);
  bf16_t* Wh_bf = (bf16_t*)p; p += (size_t)128 * 256 * sizeof(bf16_t);
  bf16_t* Wr_bf = (bf16_t*)p; p += (size_t)128 * 128 * sizeof(bf16_t);
  bf16_t* Ur_bf = (bf16_t*)p; p += (size_t)128 * 128 * sizeof(bf16_t);

  const int n_elem = NPAD * H;
  DGCNGRU_prep<<<(n_elem + 255) / 256, 256, 0, stream>>>(
      fmess, W_z, W_r, U_r, W_h, fm_bf, hA, Wz_bf, Wr_bf, Ur_bf, Wh_bf);

  DGCNGRU_r1<<<NBLK_R1, 256, 0, stream>>>(fm_bf, Wr_bf, r1);

  bf16_t* hin = hA;
  bf16_t* hout = hB;
  for (int d = 0; d < 5; d++) {
    DGCNGRU_step<<<NBLK_STEP, 256, 0, stream>>>(
        hin, hout, bgraph, fm_bf, r1, Wz_bf, Ur_bf, Wh_bf,
        b_z, b_Ur, b_h, out, (d == 4) ? 1 : 0);
    bf16_t* t = hin; hin = hout; hout = t;
  }
}
